// GlobalContextPooling_19670950216359
// MI455X (gfx1250) — matrix operand order repaired and checked by reading, not compiled, not run on hardware
//
#include <hip/hip_runtime.h>
#include <cstdint>
#include <cstddef>

// ---------------------------------------------------------------------------
// Types for CDNA5 WMMA (wave32): D = A(16x32 bf16) x B(32x16 bf16) + C(16x16 f32)
// ---------------------------------------------------------------------------
typedef __attribute__((ext_vector_type(16))) __bf16 v16bf;
typedef __attribute__((ext_vector_type(8)))  __bf16 v8bf;
typedef __attribute__((ext_vector_type(8)))  float  v8f;
typedef __attribute__((ext_vector_type(4)))  int    v4i;

#if __has_builtin(__builtin_amdgcn_global_load_async_to_lds_b128) && \
    __has_builtin(__builtin_amdgcn_s_wait_asynccnt)
#define HAVE_ASYNC_LDS 1
#endif

__device__ __forceinline__ unsigned short f32_bf16_bits(float f) {
    unsigned u = __float_as_uint(f);
    return (unsigned short)((u + 0x7FFFu + ((u >> 16) & 1u)) >> 16);
}

#define ROWH 40   // halves per LDS row: 80 bytes -> 16B aligned, bank-spread

// ---------------------------------------------------------------------------
// Implicit-GEMM conv (NHWC, bf16 WMMA, fused bias+BN+ReLU).
// Block: 256 threads = 8 waves; tile 64 pixels x 64 out-channels.
// Waves: wm = wave&3 (16-pixel M sub-tile), wn = wave>>2 (32-channel N pair).
// Each wave: 1 A fragment, 2 B fragments, 2 WMMAs per 32-channel K chunk.
// A staged via VALU f32->bf16 convert; B staged with the gfx1250 async
// global->LDS datapath when available (ASYNCcnt + s_wait_asynccnt).
// All spatial dims are powers of two: pixel decode is shift/mask only.
// trans=1 implements the stride-2 transposed conv (lhs_dilation=2, pad(2,3));
// flipped/swapped weights are folded into packing, parity-invalid taps -> 0.
// ---------------------------------------------------------------------------
__global__ __launch_bounds__(256) void conv_wmma_bf16(
    const float* __restrict__ in1, int C1,
    const float* __restrict__ in2, int C2,
    const unsigned short* __restrict__ wp,   // packed bf16 [tap][co][ci]
    const float* __restrict__ bias,
    const float* __restrict__ gamma, const float* __restrict__ beta,
    const float* __restrict__ mean,  const float* __restrict__ var,
    float* __restrict__ out,                 // NHWC
    int HI, int WI, int hwShift, int woShift, int COUT,
    int KH, int KW, int stride, int pad, int trans, int relu)
{
    __shared__ __align__(16) unsigned short As[64 * ROWH];
    __shared__ __align__(16) unsigned short Bs[64 * ROWH];

    const int CIN  = C1 + C2;
    const int tid  = threadIdx.x;
    const int lane = tid & 31;
    const int wave = tid >> 5;
    const int wm   = wave & 3;      // M sub-tile 0..3
    const int wn   = wave >> 2;     // N sub-tile 0..1 (32 channels each)
    const int mBase  = blockIdx.x * 64;
    const int coTile = blockIdx.y * 64;

    // staging role: 64 rows x (4 threads x 8 elems)
    const int sRow = tid >> 2;           // 0..63 : pixel (A) / co (B)
    const int sCol = (tid & 3) * 8;      // 0,8,16,24
    const int pixS = mBase + sRow;
    const int woMask = (1 << woShift) - 1;
    const int hoMask = (1 << (hwShift - woShift)) - 1;
    const int bS  = pixS >> hwShift;
    const int oyS = (pixS >> woShift) & hoMask;
    const int oxS = pixS & woMask;

    v8f acc0 = {0.f,0.f,0.f,0.f,0.f,0.f,0.f,0.f};
    v8f acc1 = {0.f,0.f,0.f,0.f,0.f,0.f,0.f,0.f};

    // fragment roles (wave32; contiguous k = 16*(lane/16) + 0..15 for both A and B)
    const int rowA = wm * 16 + (lane & 15);   // A: M row in LDS tile
    const int offA = (lane & 16) ? 16 : 0;    // A: K offset
    const int cw0  = wn * 32 + (lane & 15);   // B frag 0 row in LDS tile
    const int offB = (lane & 16) ? 16 : 0;    // B: K offset
    const int nChunks = CIN >> 5;
    const int nTaps   = KH * KW;

    int ky = 0, kx = 0;
    for (int tap = 0; tap < nTaps; ++tap) {
        int iy, ix; bool vs;
        if (trans) {
            // output(oy,ox) taps dilated input at (oy+ky-2, ox+kx-2); valid iff even
            int iy2 = oyS + ky - 2, ix2 = oxS + kx - 2;
            vs = ((iy2 | ix2) >= 0) && !((iy2 | ix2) & 1);
            iy = iy2 >> 1; ix = ix2 >> 1;
            vs = vs && (iy < HI) && (ix < WI);
        } else {
            iy = oyS * stride + ky - pad;
            ix = oxS * stride + kx - pad;
            vs = (iy >= 0) && (iy < HI) && (ix >= 0) && (ix < WI);
        }
        const size_t pixOff = (size_t)((bS * HI + iy) * WI + ix);
        const unsigned short* wtap = wp + (size_t)tap * COUT * CIN;

        for (int kc = 0; kc < nChunks; ++kc) {
            __syncthreads();
            // ---- stage B: 64co x 32ci bf16 (async global->LDS if available) --
            const unsigned short* wsrc =
                wtap + (size_t)(coTile + sRow) * CIN + (kc << 5) + sCol;
            unsigned short* bdst = &Bs[sRow * ROWH + sCol];
#ifdef HAVE_ASYNC_LDS
            __builtin_amdgcn_global_load_async_to_lds_b128(
                (__attribute__((address_space(1))) v4i*)wsrc,
                (__attribute__((address_space(3))) v4i*)bdst, 0, 0);
#else
            *reinterpret_cast<v8bf*>(bdst) =
                *reinterpret_cast<const v8bf*>(wsrc);
#endif
            // ---- stage A: 64px x 32ch f32 -> bf16 (overlaps async B copy) ----
            float4 v0 = make_float4(0.f, 0.f, 0.f, 0.f);
            float4 v1 = make_float4(0.f, 0.f, 0.f, 0.f);
            if (vs) {
                const int ci = (kc << 5) + sCol;
                const float* src; int ch, C;
                if (ci < C1) { src = in1; ch = ci;      C = C1; }
                else         { src = in2; ch = ci - C1; C = C2; }
                const float* pa = src + pixOff * (size_t)C + ch;
                v0 = *reinterpret_cast<const float4*>(pa);
                v1 = *reinterpret_cast<const float4*>(pa + 4);
                if (kc + 1 < nChunks) __builtin_prefetch(pa + 32, 0, 1);
            }
            unsigned short* da = &As[sRow * ROWH + sCol];
            da[0] = f32_bf16_bits(v0.x); da[1] = f32_bf16_bits(v0.y);
            da[2] = f32_bf16_bits(v0.z); da[3] = f32_bf16_bits(v0.w);
            da[4] = f32_bf16_bits(v1.x); da[5] = f32_bf16_bits(v1.y);
            da[6] = f32_bf16_bits(v1.z); da[7] = f32_bf16_bits(v1.w);
#ifdef HAVE_ASYNC_LDS
            __builtin_amdgcn_s_wait_asynccnt(0);
#endif
            __syncthreads();

            // ---- A fragment (two ds_load_b128) ----
            v8bf alo = *reinterpret_cast<const v8bf*>(&As[rowA * ROWH + offA]);
            v8bf ahi = *reinterpret_cast<const v8bf*>(&As[rowA * ROWH + offA + 8]);
            v16bf a  = __builtin_shufflevector(alo, ahi,
                        0,1,2,3,4,5,6,7,8,9,10,11,12,13,14,15);
            // ---- B fragments ----
            v8bf b0l = *reinterpret_cast<const v8bf*>(&Bs[cw0 * ROWH + offB]);
            v8bf b0h = *reinterpret_cast<const v8bf*>(&Bs[cw0 * ROWH + offB + 8]);
            v16bf b0 = __builtin_shufflevector(b0l, b0h,
                        0,1,2,3,4,5,6,7,8,9,10,11,12,13,14,15);
            v8bf b1l = *reinterpret_cast<const v8bf*>(&Bs[(cw0 + 16) * ROWH + offB]);
            v8bf b1h = *reinterpret_cast<const v8bf*>(&Bs[(cw0 + 16) * ROWH + offB + 8]);
            v16bf b1 = __builtin_shufflevector(b1l, b1h,
                        0,1,2,3,4,5,6,7,8,9,10,11,12,13,14,15);

            acc0 = __builtin_amdgcn_wmma_f32_16x16x32_bf16(
                       false, a, false, b0, (short)0, acc0, false, false);
            acc1 = __builtin_amdgcn_wmma_f32_16x16x32_bf16(
                       false, a, false, b1, (short)0, acc1, false, false);
        }
        if (++kx == KW) { kx = 0; ++ky; }
    }

    // ---- fused epilogue: bias + BN + ReLU, NHWC store (both co blocks) ----
    const int mOff = (lane & 16) ? 8 : 0;
    const int pix0 = mBase + wm * 16 + mOff;
#pragma unroll
    for (int half = 0; half < 2; ++half) {
        const int co = coTile + cw0 + half * 16;
        const v8f acc = half ? acc1 : acc0;
        const float bsv = bias ? bias[co] : 0.f;
        float sc = 1.f, sh = 0.f;
        if (gamma) {
            const float is = rsqrtf(var[co] + 1e-5f);
            sc = gamma[co] * is;
            sh = beta[co] - mean[co] * sc;
        }
#pragma unroll
        for (int r = 0; r < 8; ++r) {
            float vv = acc[r] + bsv;
            if (gamma) vv = vv * sc + sh;
            if (relu)  vv = fmaxf(vv, 0.f);
            out[(size_t)(pix0 + r) * COUT + co] = vv;
        }
    }
}

// ---------------------------------------------------------------------------
// Weight packing: OIHW f32 -> [tap][co][ci] bf16. trans=1 folds the
// transposed-conv flip+swap: wp[ky,kx,co,ci] = w[ci,co,KH-1-ky,KW-1-kx].
// ---------------------------------------------------------------------------
__global__ void pack_weights(const float* __restrict__ w,
                             unsigned short* __restrict__ wp,
                             int COUT, int CIN, int KH, int KW, int trans)
{
    const int i = blockIdx.x * blockDim.x + threadIdx.x;
    const int total = COUT * CIN * KH * KW;
    if (i >= total) return;
    const int ci  = i % CIN;
    const int t   = i / CIN;
    const int co  = t % COUT;
    const int tap = t / COUT;
    const int ky = tap / KW, kx = tap % KW;
    float v;
    if (!trans) v = w[(((size_t)co * CIN + ci) * KH + ky) * KW + kx];
    else        v = w[(((size_t)ci * COUT + co) * KH + (KH - 1 - ky)) * KW + (KW - 1 - kx)];
    wp[i] = f32_bf16_bits(v);
}

__global__ void zero_f32(float* __restrict__ p, size_t n) {
    size_t i  = (size_t)blockIdx.x * blockDim.x + threadIdx.x;
    size_t st = (size_t)gridDim.x * blockDim.x;
    for (; i < n; i += st) p[i] = 0.f;
}

// scatter sparse voxels: channel = z*16 + c, NHWC 192ch buffer
__global__ void scatter_feat(const float* __restrict__ feat,
                             const int* __restrict__ coord,
                             float* __restrict__ f)
{
    const int i = blockIdx.x * blockDim.x + threadIdx.x;
    if (i >= 200000 * 16) return;
    const int v = i >> 4, c = i & 15;
    const int x = coord[v * 4 + 0], y = coord[v * 4 + 1];
    const int z = coord[v * 4 + 2], b = coord[v * 4 + 3];
    f[((size_t)(b * 128 + y) * 128 + x) * 192 + z * 16 + c] = feat[i];
}

// align-corners bilinear 64x64 -> 128x128 into channels 128..191 of NHWC buffer
__global__ void bev_resize(const float* __restrict__ bev, float* __restrict__ f)
{
    const int i = blockIdx.x * blockDim.x + threadIdx.x;
    if (i >= 2 * 64 * 128 * 128) return;
    const int x = i & 127, y = (i >> 7) & 127, cb = (i >> 14) & 63, b = i >> 20;
    const float s = 63.f / 127.f;
    const float ys = y * s, xs = x * s;
    const int y0 = (int)ys, x0 = (int)xs;
    const int y1 = min(y0 + 1, 63), x1 = min(x0 + 1, 63);
    const float wy = ys - (float)y0, wx = xs - (float)x0;
    const float* pb = bev + (size_t)(b * 64 + cb) * 64 * 64;
    const float a00 = pb[y0 * 64 + x0], a01 = pb[y0 * 64 + x1];
    const float a10 = pb[y1 * 64 + x0], a11 = pb[y1 * 64 + x1];
    const float r0 = a00 + (a01 - a00) * wx;
    const float r1 = a10 + (a11 - a10) * wx;
    f[((size_t)(b * 128 + y) * 128 + x) * 192 + 128 + cb] = r0 + (r1 - r0) * wy;
}

__global__ void gather_sp(const float* __restrict__ O,
                          const int* __restrict__ coord,
                          float* __restrict__ sp)
{
    const int i = blockIdx.x * blockDim.x + threadIdx.x;
    if (i >= 200000 * 128) return;
    const int v = i >> 7, c = i & 127;
    const int x = coord[v * 4 + 0], y = coord[v * 4 + 1], b = coord[v * 4 + 3];
    sp[i] = O[((size_t)(b * 128 + y) * 128 + x) * 128 + c];
}

__global__ void o_transpose(const float* __restrict__ O, float* __restrict__ o)
{
    const int i = blockIdx.x * blockDim.x + threadIdx.x;
    if (i >= 2 * 128 * 128 * 128) return;
    const int x = i & 127, y = (i >> 7) & 127, c = (i >> 14) & 127, b = i >> 21;
    o[i] = O[((size_t)(b * 128 + y) * 128 + x) * 128 + c];
}

// ---------------------------------------------------------------------------
extern "C" void kernel_launch(void* const* d_in, const int* in_sizes, int n_in,
                              void* d_out, int out_size, void* d_ws, size_t ws_size,
                              hipStream_t stream)
{
    (void)in_sizes; (void)n_in; (void)out_size; (void)ws_size;

    // jax tree-leaf order (sorted keys): feat, bev, then params leaves, coord
    const float* feat = (const float*)d_in[0];
    const float* bevp = (const float*)d_in[1];
    const float* dec0_b  = (const float*)d_in[2];
    const float* dec0_w  = (const float*)d_in[3];
    const float* dec1_b  = (const float*)d_in[4];
    const float* dec1_be = (const float*)d_in[5];
    const float* dec1_ga = (const float*)d_in[6];
    const float* dec1_mn = (const float*)d_in[7];
    const float* dec1_va = (const float*)d_in[8];
    const float* dec1_w  = (const float*)d_in[9];
    const float* down_b  = (const float*)d_in[10];
    const float* down_be = (const float*)d_in[11];
    const float* down_ga = (const float*)d_in[12];
    const float* down_mn = (const float*)d_in[13];
    const float* down_va = (const float*)d_in[14];
    const float* down_w  = (const float*)d_in[15];
    const float* enc0_b  = (const float*)d_in[16];
    const float* enc0_w  = (const float*)d_in[17];
    const float* enc1_b  = (const float*)d_in[18];
    const float* enc1_be = (const float*)d_in[19];
    const float* enc1_ga = (const float*)d_in[20];
    const float* enc1_mn = (const float*)d_in[21];
    const float* enc1_va = (const float*)d_in[22];
    const float* enc1_w  = (const float*)d_in[23];
    const float* fus_b   = (const float*)d_in[24];
    const float* fus_be  = (const float*)d_in[25];
    const float* fus_ga  = (const float*)d_in[26];
    const float* fus_mn  = (const float*)d_in[27];
    const float* fus_va  = (const float*)d_in[28];
    const float* fus_w   = (const float*)d_in[29];
    const float* out_b   = (const float*)d_in[30];
    const float* out_be  = (const float*)d_in[31];
    const float* out_ga  = (const float*)d_in[32];
    const float* out_mn  = (const float*)d_in[33];
    const float* out_va  = (const float*)d_in[34];
    const float* out_w   = (const float*)d_in[35];
    const float* up_b    = (const float*)d_in[36];
    const float* up_be   = (const float*)d_in[37];
    const float* up_ga   = (const float*)d_in[38];
    const float* up_mn   = (const float*)d_in[39];
    const float* up_va   = (const float*)d_in[40];
    const float* up_w    = (const float*)d_in[41];
    const int*   coord   = (const int*)d_in[42];

    // workspace layout (floats)
    float* ws   = (float*)d_ws;
    float* bufF = ws;                     // 2*128*128*192 = 6,291,456 (concat in)
    float* bufX = bufF + 6291456;         // 8,388,608 ping
    float* bufY = bufX + 8388608;         // 8,388,608 pong
    float* bufE = bufY + 8388608;         // 4,194,304 persistent enc output
    float* bufO = bufE + 4194304;         // 4,194,304 final NHWC output
    unsigned short* wq = (unsigned short*)(bufO + 4194304);
    unsigned short* wpFus  = wq + 0;        // 192*128    = 24576
    unsigned short* wpEnc0 = wq + 24576;    // 128*128    = 16384
    unsigned short* wpEnc1 = wq + 40960;    // 128*128*25 = 409600
    unsigned short* wpDown = wq + 450560;   // 256*128*25 = 819200
    unsigned short* wpDec0 = wq + 1269760;  // 256*256    = 65536
    unsigned short* wpDec1 = wq + 1335296;  // 256*256*25 = 1638400
    unsigned short* wpUp   = wq + 2973696;  // 256*256*25 = 1638400
    unsigned short* wpOut  = wq + 4612096;  // 128*384    = 49152

    const dim3 blk(256);

    // 1) build fused input: zero + scatter + bilinear resize
    zero_f32<<<2048, blk, 0, stream>>>(bufF, (size_t)6291456);
    scatter_feat<<<(200000 * 16 + 255) / 256, blk, 0, stream>>>(feat, coord, bufF);
    bev_resize<<<(2 * 64 * 128 * 128) / 256, blk, 0, stream>>>(bevp, bufF);

    // 2) pack all weights to bf16 [tap][co][ci]
    auto packN = [&](const float* w, unsigned short* wpd, int co, int ci, int k, int tr) {
        const int tot = co * ci * k * k;
        pack_weights<<<(tot + 255) / 256, blk, 0, stream>>>(w, wpd, co, ci, k, k, tr);
    };
    packN(fus_w,  wpFus,  128, 192, 1, 0);
    packN(enc0_w, wpEnc0, 128, 128, 1, 0);
    packN(enc1_w, wpEnc1, 128, 128, 5, 0);
    packN(down_w, wpDown, 256, 128, 5, 0);
    packN(dec0_w, wpDec0, 256, 256, 1, 0);
    packN(dec1_w, wpDec1, 256, 256, 5, 0);
    packN(up_w,   wpUp,   256, 256, 5, 1);   // fold flip+transpose
    packN(out_w,  wpOut,  128, 384, 1, 0);

    // 3) conv pipeline (grid.x = pixels/64, grid.y = COUT/64)
    //    128x128 layers: hwShift=14, woShift=7 ; 64x64 layers: 12, 6
    conv_wmma_bf16<<<dim3(512, 2), blk, 0, stream>>>(
        bufF, 192, nullptr, 0, wpFus, fus_b, fus_ga, fus_be, fus_mn, fus_va,
        bufX, 128, 128, 14, 7, 128, 1, 1, 1, 0, 0, 1);
    conv_wmma_bf16<<<dim3(512, 2), blk, 0, stream>>>(
        bufX, 128, nullptr, 0, wpEnc0, enc0_b, nullptr, nullptr, nullptr, nullptr,
        bufY, 128, 128, 14, 7, 128, 1, 1, 1, 0, 0, 0);
    conv_wmma_bf16<<<dim3(512, 2), blk, 0, stream>>>(
        bufY, 128, nullptr, 0, wpEnc1, enc1_b, enc1_ga, enc1_be, enc1_mn, enc1_va,
        bufE, 128, 128, 14, 7, 128, 5, 5, 1, 2, 0, 1);
    conv_wmma_bf16<<<dim3(128, 4), blk, 0, stream>>>(
        bufE, 128, nullptr, 0, wpDown, down_b, down_ga, down_be, down_mn, down_va,
        bufX, 128, 128, 12, 6, 256, 5, 5, 2, 2, 0, 1);
    conv_wmma_bf16<<<dim3(128, 4), blk, 0, stream>>>(
        bufX, 256, nullptr, 0, wpDec0, dec0_b, nullptr, nullptr, nullptr, nullptr,
        bufY, 64, 64, 12, 6, 256, 1, 1, 1, 0, 0, 0);
    conv_wmma_bf16<<<dim3(128, 4), blk, 0, stream>>>(
        bufY, 256, nullptr, 0, wpDec1, dec1_b, dec1_ga, dec1_be, dec1_mn, dec1_va,
        bufX, 64, 64, 12, 6, 256, 5, 5, 1, 2, 0, 1);
    conv_wmma_bf16<<<dim3(512, 4), blk, 0, stream>>>(
        bufX, 256, nullptr, 0, wpUp, up_b, up_ga, up_be, up_mn, up_va,
        bufY, 64, 64, 14, 7, 256, 5, 5, 1, 0, 1, 1);
    conv_wmma_bf16<<<dim3(512, 2), blk, 0, stream>>>(
        bufE, 128, bufY, 256, wpOut, out_b, out_ga, out_be, out_mn, out_va,
        bufO, 128, 128, 14, 7, 128, 1, 1, 1, 0, 0, 1);

    // 4) outputs: sp gather (200000x128) then dense o in NCHW
    float* sp = (float*)d_out;
    float* o  = sp + (size_t)200000 * 128;
    gather_sp<<<(200000 * 128 + 255) / 256, blk, 0, stream>>>(bufO, coord, sp);
    o_transpose<<<(2 * 128 * 128 * 128) / 256, blk, 0, stream>>>(bufO, o);
}
